// MemoryAugmentedNetwork_20572893348187
// MI455X (gfx1250) — compile-verified, matrix operand order checked by reading, not run on hardware
//
#include <hip/hip_runtime.h>
#include <math.h>

// ---------------------------------------------------------------------------
// MemoryAugmentedNetwork for MI455X (gfx1250, wave32, WMMA bf16)
//   1) xg  = x @ Wih^T + (bih+bhh)            (big parallel WMMA GEMM, K=512)
//   2) LSTM recurrence (batch-split, 4 WGs, h in LDS as bf16, WMMA per step)
//   3) qk/wv projections                      (WMMA GEMM)
//   4) per-step memory scan (scores WMMA GEMM -> softmax -> read WMMA GEMM ->
//      sequential LRU write kernel), 256 steps, operands L2-resident
//   5) head: Wr (+br), ReLU(Wo1+bo1), Wo2+bo2 (WMMA GEMMs)
// GEMM waves compute 32x64 strips. The k-loop is unrolled 2x with ping-pong
// operand buffers (no register-copy rotation) so global_load_b128 latency is
// hidden behind the 8-WMMA groups without v_dual_mov overhead.
// Requires K % 64 == 0 (true for all GEMMs here: 512, 256, 16384).
// ---------------------------------------------------------------------------

typedef __attribute__((ext_vector_type(16))) __bf16 v16bf_t;
typedef __attribute__((ext_vector_type(8)))  float  v8f_t;

__device__ __forceinline__ unsigned short f2bf(float f) {
  unsigned int u = __float_as_uint(f);
  u += 0x7FFFu + ((u >> 16) & 1u);      // round-to-nearest-even
  return (unsigned short)(u >> 16);
}

union TileU { v16bf_t v; uint4 q[2]; };

// Load one 16x32 bf16 WMMA operand tile (A from row-major [M,K], or B from
// W row-major [N,K] -> computes A @ W^T). base = tile origin, ld in elements.
// Layout per CDNA5 ISA 7.12.2: lane%16 = row, lanes<16 hold K 0..7 & 16..23,
// lanes>=16 hold K 8..15 & 24..31 (two 16-byte loads per lane).
__device__ __forceinline__ v16bf_t load_tile_bf16(const unsigned short* base, int ld) {
  int l  = threadIdx.x & 31;
  int r  = l & 15;
  int kh = (l >> 4) << 3;               // 0 or 8
  const unsigned short* p = base + (size_t)r * ld + kh;
  TileU u;
  u.q[0] = *(const uint4*)(p);
  u.q[1] = *(const uint4*)(p + 16);
  return u.v;
}

__device__ __forceinline__ void prefetch_tile(const unsigned short* base, int ld) {
  int l = threadIdx.x & 31;
  const unsigned short* p = base + (size_t)(l & 15) * ld + ((l >> 4) << 3);
  __builtin_prefetch(p, 0, 1);          // -> global_prefetch_b8
}

// ---------------------------------------------------------------------------
// Generic WMMA GEMM: C[M,N] = A[M,K] @ W[N,K]^T (+bias) (+ReLU)
// One wave computes a 32x64 strip: 8 accumulator tiles. K-loop unrolled 2x
// with ping-pong buffers: while WMMAs consume buffer 0, buffer 1 loads, and
// vice versa. K must be a multiple of 64.
// ---------------------------------------------------------------------------
__global__ __launch_bounds__(256) void gemm_bf16_wmma(
    const unsigned short* __restrict__ A, int lda,
    const unsigned short* __restrict__ W, int ldw,
    const float* __restrict__ bias,
    float* __restrict__ Cf, int ldc,
    unsigned short* __restrict__ Cb, int ldcb,
    int Mt2, int Ng, int K, int relu)
{
  int wid = blockIdx.x * (blockDim.x >> 5) + (threadIdx.x >> 5);
  if (wid >= Mt2 * Ng) return;                   // wave-uniform; EXEC stays full
  int mt = wid / Ng, ng = wid % Ng;
  int row0 = mt << 5, col0 = ng << 6;

  v8f_t zero = {0.f,0.f,0.f,0.f,0.f,0.f,0.f,0.f};
  v8f_t acc[8];
#pragma unroll
  for (int j = 0; j < 8; ++j) acc[j] = zero;

  const unsigned short* A0 = A + (size_t)row0 * lda;
  const unsigned short* A1 = A0 + (size_t)16 * lda;
  const unsigned short* Wb[4];
#pragma unroll
  for (int j = 0; j < 4; ++j) Wb[j] = W + (size_t)(col0 + (j << 4)) * ldw;

  // pipeline prologue: buffer 0 holds k-step 0
  v16bf_t a0 = load_tile_bf16(A0, lda);
  v16bf_t a1 = load_tile_bf16(A1, lda);
  v16bf_t b0[4];
#pragma unroll
  for (int j = 0; j < 4; ++j) b0[j] = load_tile_bf16(Wb[j], ldw);

  for (int kb = 0; kb < K; kb += 64) {
    int k1 = kb + 32;                            // always < K (K % 64 == 0)
    int k2 = (kb + 64 < K) ? (kb + 64) : kb;     // tail reload, discarded
    if (kb + 96 < K) {                           // prefetch ~3 k-steps ahead
      prefetch_tile(A0 + kb + 96, lda);
      prefetch_tile(A1 + kb + 96, lda);
#pragma unroll
      for (int j = 0; j < 4; ++j) prefetch_tile(Wb[j] + kb + 96, ldw);
    }

    // fill buffer 1 (k1) while computing on buffer 0 (kb)
    v16bf_t c0 = load_tile_bf16(A0 + k1, lda);
    v16bf_t c1 = load_tile_bf16(A1 + k1, lda);
    v16bf_t b1[4];
#pragma unroll
    for (int j = 0; j < 4; ++j) b1[j] = load_tile_bf16(Wb[j] + k1, ldw);

#pragma unroll
    for (int j = 0; j < 4; ++j) {
      acc[j]     = __builtin_amdgcn_wmma_f32_16x16x32_bf16(
                       false, a0, false, b0[j], (short)0, acc[j],     false, false);
      acc[j + 4] = __builtin_amdgcn_wmma_f32_16x16x32_bf16(
                       false, a1, false, b0[j], (short)0, acc[j + 4], false, false);
    }

    // refill buffer 0 (k2) while computing on buffer 1 (k1)
    a0 = load_tile_bf16(A0 + k2, lda);
    a1 = load_tile_bf16(A1 + k2, lda);
#pragma unroll
    for (int j = 0; j < 4; ++j) b0[j] = load_tile_bf16(Wb[j] + k2, ldw);

#pragma unroll
    for (int j = 0; j < 4; ++j) {
      acc[j]     = __builtin_amdgcn_wmma_f32_16x16x32_bf16(
                       false, c0, false, b1[j], (short)0, acc[j],     false, false);
      acc[j + 4] = __builtin_amdgcn_wmma_f32_16x16x32_bf16(
                       false, c1, false, b1[j], (short)0, acc[j + 4], false, false);
    }
  }

  int l  = threadIdx.x & 31;
  int cn = l & 15;
  int rb = (l >> 4) << 3;
#pragma unroll
  for (int h = 0; h < 2; ++h) {
#pragma unroll
    for (int j = 0; j < 4; ++j) {
      int col = col0 + (j << 4) + cn;
      float bv = bias ? bias[col] : 0.f;
#pragma unroll
      for (int e = 0; e < 8; ++e) {
        float v = acc[h * 4 + j][e] + bv;
        if (relu) v = fmaxf(v, 0.f);
        int row = row0 + (h << 4) + rb + e;
        if (Cf) Cf[(size_t)row * ldc + col] = v;
        if (Cb) Cb[(size_t)row * ldcb + col] = f2bf(v);
      }
    }
  }
}

// ---------------------------------------------------------------------------
// LSTM recurrence. Grid = 4 blocks (16 batches each), 512 threads = 16 waves.
// Wave w owns hidden-column tile ct=w; its 4 gate tiles are col tiles
// ct, ct+16, ct+32, ct+48 of the [*,1024] gate matrix (i,f,g,o order).
// h (bf16) lives in LDS; c lives in registers. k-loop ping-pong buffered.
// ---------------------------------------------------------------------------
__global__ __launch_bounds__(512) void lstm_wmma(
    const float* __restrict__ xg,            // [B*S, 1024] = x@Wih^T + b
    const unsigned short* __restrict__ Whh,  // [1024, 256] bf16
    unsigned short* __restrict__ comb,       // [B*S, 512] bf16, write cols [0,256)
    float* __restrict__ out_h, float* __restrict__ out_c)   // [B,256]
{
  __shared__ unsigned short hls[16 * 256];
  const int S = 256, G = 1024;
  int b0 = blockIdx.x << 4;
  int ct = threadIdx.x >> 5;        // wave id = column tile
  int l  = threadIdx.x & 31;
  int cn = l & 15;
  int rb = (l >> 4) << 3;
  int hc = (ct << 4) + cn;          // hidden column

  const unsigned short* Wq[4];
#pragma unroll
  for (int q = 0; q < 4; ++q) Wq[q] = Whh + (size_t)((q << 8) + (ct << 4)) * 256;

  for (int i = threadIdx.x; i < 16 * 256; i += blockDim.x) hls[i] = 0;
  __syncthreads();

  float c[8];
#pragma unroll
  for (int e = 0; e < 8; ++e) c[e] = 0.f;

  for (int t = 0; t < S; ++t) {
    v8f_t acc[4];
#pragma unroll
    for (int q = 0; q < 4; ++q)
#pragma unroll
      for (int e = 0; e < 8; ++e) {
        int gb = b0 + rb + e;
        acc[q][e] = xg[((size_t)gb * S + t) * G + (q << 8) + hc];
      }

    v16bf_t a0 = load_tile_bf16(hls, 256);
    v16bf_t w0[4];
#pragma unroll
    for (int q = 0; q < 4; ++q) w0[q] = load_tile_bf16(Wq[q], 256);

    for (int kb = 0; kb < 256; kb += 64) {
      int k1 = kb + 32;
      int k2 = (kb + 64 < 256) ? (kb + 64) : kb;   // tail reload, discarded
      v16bf_t a1 = load_tile_bf16(hls + k1, 256);
      v16bf_t w1[4];
#pragma unroll
      for (int q = 0; q < 4; ++q) w1[q] = load_tile_bf16(Wq[q] + k1, 256);
#pragma unroll
      for (int q = 0; q < 4; ++q)
        acc[q] = __builtin_amdgcn_wmma_f32_16x16x32_bf16(
                     false, a0, false, w0[q], (short)0, acc[q], false, false);
      a0 = load_tile_bf16(hls + k2, 256);
#pragma unroll
      for (int q = 0; q < 4; ++q) w0[q] = load_tile_bf16(Wq[q] + k2, 256);
#pragma unroll
      for (int q = 0; q < 4; ++q)
        acc[q] = __builtin_amdgcn_wmma_f32_16x16x32_bf16(
                     false, a1, false, w1[q], (short)0, acc[q], false, false);
    }

    float hv[8];
#pragma unroll
    for (int e = 0; e < 8; ++e) {
      float ig = 1.f / (1.f + __expf(-acc[0][e]));
      float fg = 1.f / (1.f + __expf(-acc[1][e]));
      float gg = tanhf(acc[2][e]);
      float og = 1.f / (1.f + __expf(-acc[3][e]));
      c[e]  = fg * c[e] + ig * gg;
      hv[e] = og * tanhf(c[e]);
    }

    __syncthreads();                 // all waves done reading old h
#pragma unroll
    for (int e = 0; e < 8; ++e) {
      unsigned short hb = f2bf(hv[e]);
      hls[(rb + e) * 256 + hc] = hb;
      int gb = b0 + rb + e;
      comb[((size_t)gb * S + t) * 512 + hc] = hb;
    }
    __syncthreads();                 // new h visible before next step's WMMA

    if (t == S - 1) {
#pragma unroll
      for (int e = 0; e < 8; ++e) {
        int gb = b0 + rb + e;
        out_h[(size_t)gb * 256 + hc] = hv[e];
        out_c[(size_t)gb * 256 + hc] = c[e];
      }
    }
  }
}

// ---------------------------------------------------------------------------
// Row softmax over M=16384 columns, bf16 output. One block per row.
// ---------------------------------------------------------------------------
__global__ __launch_bounds__(256) void softmax_bf16(
    const float* __restrict__ sc, unsigned short* __restrict__ attn, int Mc)
{
  int row = blockIdx.x;
  const float* p = sc + (size_t)row * Mc;
  __shared__ float red[256];
  int tid = threadIdx.x;

  float mx = -3.4e38f;
  for (int i = tid; i < Mc; i += 256) mx = fmaxf(mx, p[i]);
  red[tid] = mx; __syncthreads();
  for (int s = 128; s > 0; s >>= 1) {
    if (tid < s) red[tid] = fmaxf(red[tid], red[tid + s]);
    __syncthreads();
  }
  float m = red[0]; __syncthreads();

  float sum = 0.f;
  for (int i = tid; i < Mc; i += 256) sum += __expf(p[i] - m);
  red[tid] = sum; __syncthreads();
  for (int s = 128; s > 0; s >>= 1) {
    if (tid < s) red[tid] += red[tid + s];
    __syncthreads();
  }
  float inv = 1.f / red[0];

  for (int i = tid; i < Mc; i += 256)
    attn[(size_t)row * Mc + i] = f2bf(__expf(p[i] - m) * inv);
}

// ---------------------------------------------------------------------------
// Sequential LRU memory writes for one timestep: for b = 0..63 (in order):
//   lru = first argmin(usage); memK[lru,:]=qk[b,t,:]; memV^T[:,lru]=wv[b,t,:];
//   usage[lru] = max(usage)+1.  Single workgroup, 1024-thread reductions.
// ---------------------------------------------------------------------------
__global__ __launch_bounds__(1024) void mem_write_kernel(
    float* __restrict__ usage,
    unsigned short* __restrict__ memK,    // [M, 256] bf16
    unsigned short* __restrict__ memVT,   // [256, M] bf16 (transposed values)
    const float* __restrict__ qk,         // [B*S, 256] f32
    const float* __restrict__ wv,         // [B*S, 256] f32
    int t)
{
  const int MM = 16384, S = 256, D = 256;
  __shared__ float rmin[1024];
  __shared__ int   ridx[1024];
  __shared__ float rmax[1024];
  __shared__ int   s_lru;
  int tid = threadIdx.x;

  for (int b = 0; b < 64; ++b) {
    float mn = 3.4e38f; int mi = 0x7fffffff; float mxv = -3.4e38f;
    for (int i = tid; i < MM; i += 1024) {
      float u = usage[i];
      if (u < mn) { mn = u; mi = i; }     // ascending scan -> first-min per thread
      mxv = fmaxf(mxv, u);
    }
    rmin[tid] = mn; ridx[tid] = mi; rmax[tid] = mxv;
    __syncthreads();
    for (int s = 512; s > 0; s >>= 1) {
      if (tid < s) {
        if (rmin[tid + s] < rmin[tid] ||
            (rmin[tid + s] == rmin[tid] && ridx[tid + s] < ridx[tid])) {
          rmin[tid] = rmin[tid + s]; ridx[tid] = ridx[tid + s];
        }
        rmax[tid] = fmaxf(rmax[tid], rmax[tid + s]);
      }
      __syncthreads();
    }
    if (tid == 0) {
      s_lru = ridx[0];
      usage[ridx[0]] = rmax[0] + 1.0f;
    }
    __syncthreads();
    int lru = s_lru;
    if (tid < D) {
      size_t src = ((size_t)b * S + t) * D + tid;
      memK[(size_t)lru * D + tid]   = f2bf(qk[src]);
      memVT[(size_t)tid * MM + lru] = f2bf(wv[src]);
    }
    __threadfence();
    __syncthreads();
  }
}

// ---------------------------------------------------------------------------
// Small helpers
// ---------------------------------------------------------------------------
__global__ void cvt_bf16_kernel(const float* __restrict__ in,
                                unsigned short* __restrict__ out, int n) {
  int i = blockIdx.x * blockDim.x + threadIdx.x;
  if (i < n) out[i] = f2bf(in[i]);
}
__global__ void transpose_bf16_kernel(const float* __restrict__ in,     // [Mr, Dc]
                                      unsigned short* __restrict__ out, // [Dc, Mr]
                                      int Mr, int Dc) {
  int i = blockIdx.x * blockDim.x + threadIdx.x;
  if (i < Mr * Dc) {
    int m = i / Dc, d = i % Dc;
    out[(size_t)d * Mr + m] = f2bf(in[i]);
  }
}
__global__ void copy_f32_kernel(const float* __restrict__ in,
                                float* __restrict__ out, int n) {
  int i = blockIdx.x * blockDim.x + threadIdx.x;
  if (i < n) out[i] = in[i];
}
__global__ void add2_kernel(const float* __restrict__ a, const float* __restrict__ b,
                            float* __restrict__ o, int n) {
  int i = blockIdx.x * blockDim.x + threadIdx.x;
  if (i < n) o[i] = a[i] + b[i];
}

// ---------------------------------------------------------------------------
// Host orchestration
// ---------------------------------------------------------------------------
extern "C" void kernel_launch(void* const* d_in, const int* in_sizes, int n_in,
                              void* d_out, int out_size, void* d_ws, size_t ws_size,
                              hipStream_t stream)
{
  (void)in_sizes; (void)n_in; (void)out_size; (void)ws_size;
  constexpr int B = 64, S = 256, I = 512, H = 256, G = 1024, D = 256, MM = 16384, O = 64;

  const float* x   = (const float*)d_in[0];
  const float* Wih = (const float*)d_in[1];
  const float* Whh = (const float*)d_in[2];
  const float* bih = (const float*)d_in[3];
  const float* bhh = (const float*)d_in[4];
  const float* Wk  = (const float*)d_in[5];
  const float* bk  = (const float*)d_in[6];
  const float* Wv  = (const float*)d_in[7];
  const float* bv  = (const float*)d_in[8];
  const float* Wr  = (const float*)d_in[9];
  const float* br  = (const float*)d_in[10];
  const float* Wo1 = (const float*)d_in[11];
  const float* bo1 = (const float*)d_in[12];
  const float* Wo2 = (const float*)d_in[13];
  const float* bo2 = (const float*)d_in[14];
  const float* mk0 = (const float*)d_in[15];
  const float* mv0 = (const float*)d_in[16];
  const float* mu0 = (const float*)d_in[17];
  float* out = (float*)d_out;

  char* wp = (char*)d_ws;
  auto alloc = [&](size_t bytes) {
    char* p = wp; wp += (bytes + 255) & ~(size_t)255; return p;
  };

  unsigned short* x_bf    = (unsigned short*)alloc((size_t)B * S * I * 2);
  unsigned short* Wih_bf  = (unsigned short*)alloc((size_t)G * I * 2);
  unsigned short* Whh_bf  = (unsigned short*)alloc((size_t)G * H * 2);
  unsigned short* Wk_bf   = (unsigned short*)alloc((size_t)D * H * 2);
  unsigned short* Wv_bf   = (unsigned short*)alloc((size_t)D * H * 2);
  unsigned short* Wr_bf   = (unsigned short*)alloc((size_t)H * (H + D) * 2);
  unsigned short* Wo1_bf  = (unsigned short*)alloc((size_t)H * H * 2);
  unsigned short* Wo2_bf  = (unsigned short*)alloc((size_t)O * H * 2);
  float*          bsum    = (float*)alloc((size_t)G * 4);
  float*          xg      = (float*)alloc((size_t)B * S * G * 4);
  unsigned short* comb    = (unsigned short*)alloc((size_t)B * S * (H + D) * 2);
  float*          qk_f    = (float*)alloc((size_t)B * S * D * 4);
  unsigned short* qk_bf   = (unsigned short*)alloc((size_t)B * S * D * 2);
  float*          wv_f    = (float*)alloc((size_t)B * S * D * 4);
  unsigned short* memK_bf = (unsigned short*)alloc((size_t)MM * D * 2);
  unsigned short* memVT   = (unsigned short*)alloc((size_t)D * MM * 2);
  float*          usage   = (float*)alloc((size_t)MM * 4);
  float*          scores  = (float*)alloc((size_t)B * MM * 4);
  unsigned short* attn_bf = (unsigned short*)alloc((size_t)B * MM * 2);
  unsigned short* ro_bf   = (unsigned short*)alloc((size_t)B * S * H * 2);
  unsigned short* r1_bf   = (unsigned short*)alloc((size_t)B * S * H * 2);

  auto cvt = [&](const float* src, unsigned short* dst, int n) {
    cvt_bf16_kernel<<<(n + 255) / 256, 256, 0, stream>>>(src, dst, n);
  };
  auto gemm = [&](const unsigned short* A, int lda, const unsigned short* W_, int ldw,
                  const float* bias, float* Cf, int ldc, unsigned short* Cb, int ldcb,
                  int Mr, int N, int K, int relu) {
    int Mt2 = Mr >> 5, Ng = N >> 6;          // 32-row, 64-col strips per wave
    int blocks = (Mt2 * Ng + 7) / 8;
    gemm_bf16_wmma<<<blocks, 256, 0, stream>>>(A, lda, W_, ldw, bias,
                                               Cf, ldc, Cb, ldcb, Mt2, Ng, K, relu);
  };

  // --- input conversion / init (every call; deterministic across replays) ---
  cvt(x, x_bf, B * S * I);
  cvt(Wih, Wih_bf, G * I);
  cvt(Whh, Whh_bf, G * H);
  cvt(Wk, Wk_bf, D * H);
  cvt(Wv, Wv_bf, D * H);
  cvt(Wr, Wr_bf, H * (H + D));
  cvt(Wo1, Wo1_bf, H * H);
  cvt(Wo2, Wo2_bf, O * H);
  cvt(mk0, memK_bf, MM * D);
  transpose_bf16_kernel<<<(MM * D + 255) / 256, 256, 0, stream>>>(mv0, memVT, MM, D);
  copy_f32_kernel<<<(MM + 255) / 256, 256, 0, stream>>>(mu0, usage, MM);
  add2_kernel<<<(G + 255) / 256, 256, 0, stream>>>(bih, bhh, bsum, G);

  // --- 1) xg = x @ Wih^T + (bih+bhh), rows = b*S+t ---
  gemm(x_bf, I, Wih_bf, I, bsum, xg, G, nullptr, 0, B * S, G, I, 0);

  // --- 2) LSTM recurrence (writes comb cols [0,H) and final h,c) ---
  float* out_h = out + (size_t)B * S * O;
  float* out_c = out_h + (size_t)B * H;
  lstm_wmma<<<4, 512, 0, stream>>>(xg, Whh_bf, comb, out_h, out_c);

  // --- 3) qk / wv projections (A = comb cols [0,H), lda = H+D) ---
  gemm(comb, H + D, Wk_bf, H, bk, qk_f, D, qk_bf, D, B * S, D, H, 0);
  gemm(comb, H + D, Wv_bf, H, bv, wv_f, D, nullptr, 0, B * S, D, H, 0);

  // --- 4) memory scan: 256 sequential steps ---
  for (int t = 0; t < S; ++t) {
    // scores[64, M] = Q_t @ memK^T   (A row b at qk_bf + (b*S+t)*D)
    gemm(qk_bf + (size_t)t * D, S * D, memK_bf, D, nullptr,
         scores, MM, nullptr, 0, B, MM, D, 0);
    softmax_bf16<<<B, 256, 0, stream>>>(scores, attn_bf, MM);
    // reads_t[64, D] = attn @ memV = attn @ memVT^T -> comb cols [H, H+D)
    gemm(attn_bf, MM, memVT, MM, nullptr,
         nullptr, 0, comb + (size_t)t * (H + D) + H, S * (H + D), B, D, MM, 0);
    // sequential LRU writes for this step
    mem_write_kernel<<<1, 1024, 0, stream>>>(usage, memK_bf, memVT, qk_f, wv_f, t);
  }

  // --- 5) head ---
  gemm(comb, H + D, Wr_bf, H + D, br, nullptr, 0, ro_bf, H, B * S, H, H + D, 0);
  gemm(ro_bf, H, Wo1_bf, H, bo1, nullptr, 0, r1_bf, H, B * S, H, H, 1);
  gemm(r1_bf, H, Wo2_bf, H, bo2, out, O, nullptr, 0, B * S, O, H, 0);
}
